// Swin_v2_4947802325770
// MI455X (gfx1250) — matrix operand order repaired and checked by reading, not compiled, not run on hardware
//
#include <hip/hip_runtime.h>
#include <hip/hip_bf16.h>

typedef __bf16 bf16;
typedef __attribute__((ext_vector_type(16))) __bf16 v16bf;
typedef __attribute__((ext_vector_type(8)))  __bf16 v8bf;
typedef __attribute__((ext_vector_type(8)))  float  v8f;

#define DIM   96
#define HEADS 6
#define NWIN  800
#define NTOK  147
#define NPAD  160
#define QSCALE 0.25f      // (96/6)^-0.5
#define LNEPS 1e-5f

// ================= WMMA operand swizzle (CDNA5 ISA 7.12.2, wave32) =================
// A fragment element (m,k) of a 16x32 tile: lane = m%16 + 16*((k>>3)&1),
// reg element e = (k&7) + 8*((k>>4)&1).  Stored pre-swizzled: tile = 32 lanes x 16 bf16.
__device__ __forceinline__ long aswIndex(long mTile, int kt, int mr, int kin) {
  int lane = mr + ((kin & 8) ? 16 : 0);
  int e = (kin & 7) + ((kin & 16) ? 8 : 0);
  return ((mTile * 3 + kt) * 32 + lane) * 16 + e;
}
// B fragment element (k,n) of a 32x16 tile: lane = n%16 + 16*((k>>3)&1), e = (k&7) + 8*((k>>4)&1).
__device__ __forceinline__ long bswIndex(int kt, int nt, int nTiles, int kin, int nr) {
  int lane = nr + ((kin & 8) ? 16 : 0);
  int e = (kin & 7) + ((kin & 16) ? 8 : 0);
  return (((long)kt * nTiles + nt) * 32 + lane) * 16 + e;
}
// one aligned 32B load per operand (2x global_load_b128)
__device__ __forceinline__ v16bf loadSw(const bf16* base, long tileIdx, int lane) {
  return *(const v16bf*)(base + (tileIdx * 32 + lane) * 16);
}

// 1 M-tile x 3 N-tiles, K=96 (3 WMMA k-steps); A loaded once per k-step.
__device__ __forceinline__ void gemm3(const bf16* ASW, const bf16* BSW, int nTiles,
                                      long mTile, int g, int lane, v8f acc[3]) {
  for (int kt = 0; kt < 3; ++kt) {
    v16bf a = loadSw(ASW, mTile * 3 + kt, lane);
#pragma unroll
    for (int j = 0; j < 3; ++j) {
      v16bf b = loadSw(BSW, (long)kt * nTiles + g * 3 + j, lane);
      acc[j] = __builtin_amdgcn_wmma_f32_16x16x32_bf16(false, a, false, b, (short)0,
                                                       acc[j], false, false);
    }
  }
}

// ---------------- 0a: weight prep (fp32 -> bf16, pre-swizzled B fragments) -----------
__global__ void k_prep(const float* wqkv, const float* wproj, const float* wpw,
                       bf16* wqkvT, bf16* wprojT, bf16* wpwT) {
  int t = blockIdx.x * blockDim.x + threadIdx.x;
  if (t < 96 * 288) {
    int k = t / 288, n = t % 288;
    wqkvT[bswIndex(k >> 5, n >> 4, 18, k & 31, n & 15)] = (bf16)wqkv[n * 96 + k];
    return;
  }
  t -= 96 * 288;
  if (t < 96 * 96) {
    int k = t / 96, n = t % 96;
    wprojT[bswIndex(k >> 5, n >> 4, 6, k & 31, n & 15)] = (bf16)wproj[n * 96 + k];
    return;
  }
  t -= 96 * 96;
  if (t < 96 * 96) {
    int k = t / 96, n = t % 96;
    wpwT[bswIndex(k >> 5, n >> 4, 6, k & 31, n & 15)] = (bf16)wpw[n * 96 + k];
  }
}

// ------- 0b: attention tables: rel-pos bias (head,i,j) and shift-mask category -------
__global__ void k_tabs(const float* rpb, float* bias, signed char* cat) {
  int t = blockIdx.x * blockDim.x + threadIdx.x;
  if (t < HEADS * NTOK * NTOK) {
    int head = t / (NTOK * NTOK);
    int rem = t % (NTOK * NTOK);
    int i = rem / NTOK, j = rem % NTOK;
    int tdi = i / 49, thi = (i % 49) / 7, twi = i % 7;
    int tdj = j / 49, thj = (j % 49) / 7, twj = j % 7;
    int idx = (tdi - tdj + 2) * 169 + (thi - thj + 6) * 13 + (twi - twj + 6);
    bias[t] = rpb[idx * HEADS + head];
    return;
  }
  t -= HEADS * NTOK * NTOK;
  if (t < NWIN * NPAD) {
    int win = t / NPAD, tok = t % NPAD;
    signed char c = 127;                       // padded tokens: unique category
    if (tok < NTOK) {
      int wd = win / 400, wh = (win % 400) / 20, ww = win % 20;
      int td = tok / 49, th = (tok % 49) / 7, tw = tok % 7;
      int d = wd * 3 + td, h = wh * 7 + th, w = ww * 7 + tw;
      c = (signed char)((d < 3 ? 0 : (d < 5 ? 1 : 2)) * 9 +
                        (h < 133 ? 0 : (h < 137 ? 1 : 2)) * 3 +
                        (w < 133 ? 0 : (w < 137 ? 1 : 2)));
    }
    cat[t] = c;
  }
}

// ------- 1: LN1 + cyclic shift + window partition -> A-swizzled bf16 (pad rows 0) ----
__global__ void k_ln_shift(const float* x, const float* g1, const float* b1, bf16* xw) {
  int t = blockIdx.x * blockDim.x + threadIdx.x;   // m = win*160 + row
  if (t >= NWIN * NPAD) return;
  int win = t / NPAD, row = t % NPAD;
  long mTile = t >> 4;
  int  mr = t & 15;
  if (row >= NTOK) {
    for (int c = 0; c < DIM; ++c) xw[aswIndex(mTile, c >> 5, mr, c & 31)] = (bf16)0.0f;
    return;
  }
  int wd = win / 400, wh = (win % 400) / 20, ww = win % 20;
  int td = row / 49, th = (row % 49) / 7, tw = row % 7;
  int sd = (wd * 3 + td + 1) % 6;       // roll -1
  int sh = (wh * 7 + th + 3) % 140;     // roll -3
  int sw = (ww * 7 + tw + 3) % 140;
  const float4* s4 = (const float4*)(x + (((long)sd * 140 + sh) * 140 + sw) * DIM);
  float buf[DIM];
  float mu = 0.f;
#pragma unroll
  for (int q = 0; q < DIM / 4; ++q) {
    float4 v = s4[q];
    buf[q * 4 + 0] = v.x; buf[q * 4 + 1] = v.y; buf[q * 4 + 2] = v.z; buf[q * 4 + 3] = v.w;
    mu += v.x + v.y + v.z + v.w;
  }
  mu *= (1.0f / DIM);
  float var = 0.f;
  for (int c = 0; c < DIM; ++c) { float d = buf[c] - mu; var += d * d; }
  float rs = rsqrtf(var * (1.0f / DIM) + LNEPS);
  for (int c = 0; c < DIM; ++c)
    xw[aswIndex(mTile, c >> 5, mr, c & 31)] = (bf16)((buf[c] - mu) * rs * g1[c] + b1[c]);
}

// ---------------- 2: QKV GEMM (M=128000, K=96, N=288), q pre-scaled ----------------
__global__ void k_qkv(const bf16* xw, const bf16* wqkvT, const float* bqkv, bf16* qkv) {
  int lane = threadIdx.x & 31, wave = threadIdx.x >> 5;
  long task = (long)blockIdx.x * 8 + wave;
  if (task >= 8000L * 6) return;
  int  g = (int)(task % 6);
  long tm = task / 6;
  v8f acc[3] = {};
  gemm3(xw, wqkvT, 18, tm, g, lane, acc);
#pragma unroll
  for (int j = 0; j < 3; ++j) {
    int n = (g * 3 + j) * 16 + (lane & 15);
    float bias = bqkv[n];
    float sc = (n < 96) ? QSCALE : 1.0f;
#pragma unroll
    for (int r = 0; r < 8; ++r) {
      long m = tm * 16 + r + ((lane < 16) ? 0 : 8);
      qkv[m * 288 + n] = (bf16)((acc[j][r] + bias) * sc);
    }
  }
}

// --------- 3: fused window attention, LDS-staged; block = (win, head, 32-row slab) ---
__global__ void __launch_bounds__(64) k_attn(const bf16* qkv, const float* bias,
                                             const signed char* cat, bf16* ao) {
  __shared__ float S[32 * NPAD];     // scores fp32
  __shared__ bf16  P[32 * NPAD];     // probabilities bf16
  __shared__ bf16  Qs[32 * 32];      // 32 q-rows, K padded 16->32 with zeros
  __shared__ bf16  Ks[NPAD * 32];    // 160 k-rows, K padded 16->32 with zeros
  __shared__ bf16  VT[16 * NPAD];    // v transposed: [d][token]
  int blk  = blockIdx.x;             // win*30 + head*5 + rb
  int win  = blk / 30;
  int head = (blk % 30) / 5;
  int rb   = blk % 5;
  int lane = threadIdx.x & 31, wave = threadIdx.x >> 5;
  int tid  = threadIdx.x;
  int rowBase = rb * 32;
  const bf16* qk = qkv + (long)win * NPAD * 288;
  const signed char* catw = cat + win * NPAD;
  const float* biash = bias + (long)head * NTOK * NTOK;

  // ---- stage K (zero-padded) and Q via 16B chunks; V transposed scalar ----
  uint4 zz; zz.x = zz.y = zz.z = zz.w = 0u;
  for (int u = tid; u < NPAD * 2; u += 64) {           // K rows: 2 chunks of 8 bf16
    int r = u >> 1, h2 = u & 1;
    *(uint4*)&Ks[r * 32 + h2 * 8] =
        *(const uint4*)&qk[(long)r * 288 + 96 + head * 16 + h2 * 8];
    *(uint4*)&Ks[r * 32 + 16 + h2 * 8] = zz;
  }
  {
    int u = tid;                                        // 64 chunks exactly for Q
    int r = u >> 1, h2 = u & 1;
    *(uint4*)&Qs[r * 32 + h2 * 8] =
        *(const uint4*)&qk[(long)(rowBase + r) * 288 + head * 16 + h2 * 8];
    *(uint4*)&Qs[r * 32 + 16 + h2 * 8] = zz;
  }
  for (int u = tid; u < 16 * NPAD; u += 64) {           // V transpose
    int d = u / NPAD, tok = u % NPAD;
    VT[u] = qk[(long)tok * 288 + 192 + head * 16 + d];
  }
  __syncthreads();

  // ---- phase 1: S = q @ k^T (scale folded into q), 2x10 tiles ----
  for (int t = wave; t < 20; t += 2) {
    int ti = t / 10, tj = t % 10;
    const bf16* qrow = Qs + (ti * 16 + (lane & 15)) * 32 + ((lane < 16) ? 0 : 16);
    v8bf alo = *(const v8bf*)qrow;
    v8bf ahi = *(const v8bf*)(qrow + 8);
    v16bf a = __builtin_shufflevector(alo, ahi, 0, 1, 2, 3, 4, 5, 6, 7,
                                      8, 9, 10, 11, 12, 13, 14, 15);
    v16bf b = *(const v16bf*)&Ks[(tj * 16 + (lane & 15)) * 32 + ((lane < 16) ? 0 : 16)];
    v8f c = {};
    c = __builtin_amdgcn_wmma_f32_16x16x32_bf16(false, a, false, b, (short)0, c,
                                                false, false);
    int j = tj * 16 + (lane & 15);
    signed char cj = (j < NPAD) ? catw[j] : (signed char)127;
#pragma unroll
    for (int r = 0; r < 8; ++r) {
      int iLoc = ti * 16 + r + ((lane < 16) ? 0 : 8);
      int i = rowBase + iLoc;
      float v;
      if (j >= NTOK)      v = -1.0e30f;
      else if (i >= NTOK) v = 0.0f;
      else {
        float bs = biash[i * NTOK + j];
        v = c[r] + bs + ((catw[i] != cj) ? -100.0f : 0.0f);
      }
      S[iLoc * NPAD + j] = v;
    }
  }
  __syncthreads();

  // ---- phase 2: row softmax (32 rows, one per thread of first 32) ----
  if (tid < 32) {
    float* srow = S + tid * NPAD;
    float mx = -1.0e30f;
    for (int j = 0; j < NTOK; ++j) mx = fmaxf(mx, srow[j]);
    float sum = 0.f;
    for (int j = 0; j < NTOK; ++j) { float e = __expf(srow[j] - mx); srow[j] = e; sum += e; }
    float inv = 1.0f / sum;
    bf16* prow = P + tid * NPAD;
    for (int j = 0; j < NTOK; ++j) prow[j] = (bf16)(srow[j] * inv);
    for (int j = NTOK; j < NPAD; ++j) prow[j] = (bf16)0.0f;
  }
  __syncthreads();

  // ---- phase 3: O = P @ V, one 16-row tile per wave, K=160 = 5 WMMA steps ----
  {
    int ti = wave;
    v8f acc = {};
    for (int kb = 0; kb < NPAD; kb += 32) {
      const bf16* prow = P + (ti * 16 + (lane & 15)) * NPAD + kb + ((lane < 16) ? 0 : 16);
      v8bf alo = *(const v8bf*)prow;
      v8bf ahi = *(const v8bf*)(prow + 8);
      v16bf a = __builtin_shufflevector(alo, ahi, 0, 1, 2, 3, 4, 5, 6, 7,
                                        8, 9, 10, 11, 12, 13, 14, 15);
      v16bf b = *(const v16bf*)&VT[(lane & 15) * NPAD + kb + ((lane < 16) ? 0 : 16)];
      acc = __builtin_amdgcn_wmma_f32_16x16x32_bf16(false, a, false, b, (short)0, acc,
                                                    false, false);
    }
    // store into A-swizzled AO for the proj GEMM: k = head*16 + n
    int n = lane & 15;
    int kt = head >> 1;
    int e = (n & 7) + ((head & 1) ? 8 : 0);
    int lane2base = ((n & 8) ? 16 : 0);
#pragma unroll
    for (int r = 0; r < 8; ++r) {
      int row = rowBase + ti * 16 + r + ((lane < 16) ? 0 : 8);
      long mTile = (long)win * 10 + (row >> 4);
      ao[((mTile * 3 + kt) * 32 + (row & 15) + lane2base) * 16 + e] = (bf16)acc[r];
    }
  }
}

// ---------------- 4: proj GEMM + window-reverse + un-shift + residual ----------------
__global__ void k_proj(const bf16* ao, const bf16* wprojT, const float* bproj,
                       const float* x, float* xres) {
  int lane = threadIdx.x & 31, wave = threadIdx.x >> 5;
  long task = (long)blockIdx.x * 8 + wave;
  if (task >= 8000L * 2) return;
  int  g = (int)(task % 2);
  long tm = task / 2;
  v8f acc[3] = {};
  gemm3(ao, wprojT, 6, tm, g, lane, acc);
#pragma unroll
  for (int j = 0; j < 3; ++j) {
    int n = (g * 3 + j) * 16 + (lane & 15);
    float bias = bproj[n];
#pragma unroll
    for (int r = 0; r < 8; ++r) {
      long m = tm * 16 + r + ((lane < 16) ? 0 : 8);
      int win = (int)(m / NPAD), row = (int)(m % NPAD);
      if (row >= NTOK) continue;
      int wd = win / 400, wh = (win % 400) / 20, ww = win % 20;
      int td = row / 49, th = (row % 49) / 7, tw = row % 7;
      int d = (wd * 3 + td + 1) % 6;
      int h = (wh * 7 + th + 3) % 140;
      int w = (ww * 7 + tw + 3) % 140;
      long oi = (((long)d * 140 + h) * 140 + w) * DIM + n;
      xres[oi] = x[oi] + acc[j][r] + bias;
    }
  }
}

// ---------------- 5: LN2 -> A-swizzled bf16 ----------------
__global__ void k_ln2(const float* xres, const float* g2, const float* b2, bf16* yn) {
  int t = blockIdx.x * blockDim.x + threadIdx.x;
  if (t >= 117600) return;
  const float4* s4 = (const float4*)(xres + (long)t * DIM);
  long mTile = t >> 4;
  int  mr = t & 15;
  float buf[DIM];
  float mu = 0.f;
#pragma unroll
  for (int q = 0; q < DIM / 4; ++q) {
    float4 v = s4[q];
    buf[q * 4 + 0] = v.x; buf[q * 4 + 1] = v.y; buf[q * 4 + 2] = v.z; buf[q * 4 + 3] = v.w;
    mu += v.x + v.y + v.z + v.w;
  }
  mu *= (1.0f / DIM);
  float var = 0.f;
  for (int c = 0; c < DIM; ++c) { float d = buf[c] - mu; var += d * d; }
  float rs = rsqrtf(var * (1.0f / DIM) + LNEPS);
  for (int c = 0; c < DIM; ++c)
    yn[aswIndex(mTile, c >> 5, mr, c & 31)] = (bf16)((buf[c] - mu) * rs * g2[c] + b2[c]);
}

// ---------------- 6: pointwise conv GEMM (M=117600, K=96, N=96) ----------------
__global__ void k_pw(const bf16* yn, const bf16* wpwT, const float* bpw, bf16* y1) {
  int lane = threadIdx.x & 31, wave = threadIdx.x >> 5;
  long task = (long)blockIdx.x * 8 + wave;
  if (task >= 7350L * 2) return;
  int  g = (int)(task % 2);
  long tm = task / 2;
  v8f acc[3] = {};
  gemm3(yn, wpwT, 6, tm, g, lane, acc);
#pragma unroll
  for (int j = 0; j < 3; ++j) {
    int n = (g * 3 + j) * 16 + (lane & 15);
    float bias = bpw[n];
#pragma unroll
    for (int r = 0; r < 8; ++r) {
      long m = tm * 16 + r + ((lane < 16) ? 0 : 8);
      y1[m * DIM + n] = (bf16)(acc[j][r] + bias);
    }
  }
}

// ------- 7: depthwise 3x3x3 conv + bias + residual; channel pairs, dword taps -------
__global__ void k_dw(const bf16* y1, const float* wdw, const float* bdw,
                     const float* xres, float* out) {
  long t = (long)blockIdx.x * blockDim.x + threadIdx.x;   // (spatial, channel-pair)
  if (t >= 5644800L) return;
  int c = (int)(t % 48) * 2;
  long s = t / 48;
  int w = (int)(s % 140); s /= 140;
  int h = (int)(s % 140); s /= 140;
  int d = (int)s;
  float acc0 = bdw[c], acc1 = bdw[c + 1];
  const float* wk0 = wdw + c * 27;
  const float* wk1 = wk0 + 27;
  for (int kd = 0; kd < 3; ++kd) {
    int dd = d + kd - 1; if (dd < 0 || dd >= 6) continue;
    for (int kh = 0; kh < 3; ++kh) {
      int hh = h + kh - 1; if (hh < 0 || hh >= 140) continue;
      for (int kw = 0; kw < 3; ++kw) {
        int ww = w + kw - 1; if (ww < 0 || ww >= 140) continue;
        const bf16* yy = &y1[(((long)dd * 140 + hh) * 140 + ww) * DIM + c];
        int kk = kd * 9 + kh * 3 + kw;
        acc0 += wk0[kk] * (float)yy[0];
        acc1 += wk1[kk] * (float)yy[1];
      }
    }
  }
  long oi = ((((long)d * 140 + h) * 140 + w) * DIM) + c;
  const float2 xr = *(const float2*)&xres[oi];
  float2 o; o.x = xr.x + acc0; o.y = xr.y + acc1;
  *(float2*)&out[oi] = o;
}

extern "C" void kernel_launch(void* const* d_in, const int* in_sizes, int n_in,
                              void* d_out, int out_size, void* d_ws, size_t ws_size,
                              hipStream_t stream) {
  (void)in_sizes; (void)n_in; (void)out_size; (void)ws_size;
  const float* x     = (const float*)d_in[0];
  const float* g1    = (const float*)d_in[1];
  const float* b1    = (const float*)d_in[2];
  const float* wqkv  = (const float*)d_in[3];
  const float* bqkv  = (const float*)d_in[4];
  const float* rpb   = (const float*)d_in[5];
  const float* wproj = (const float*)d_in[6];
  const float* bproj = (const float*)d_in[7];
  const float* g2    = (const float*)d_in[8];
  const float* b2    = (const float*)d_in[9];
  const float* wpw   = (const float*)d_in[10];
  const float* bpw   = (const float*)d_in[11];
  const float* wdw   = (const float*)d_in[12];
  const float* bdw   = (const float*)d_in[13];

  char* ws = (char*)d_ws;
  bf16*  WQKVT  = (bf16*)(ws + 0);            //  55,296 B  (B-swizzled)
  bf16*  WPROJT = (bf16*)(ws + 55296);        //  18,432 B  (B-swizzled)
  bf16*  WPWT   = (bf16*)(ws + 73728);        //  18,432 B  (B-swizzled)
  bf16*  XW     = (bf16*)(ws + 92160);        //  24,576,000 B (A-swizzled)
  bf16*  QKV    = (bf16*)(ws + 24668160);     //  73,728,000 B (row-major [m][288])
  bf16*  AO     = (bf16*)(ws + 98396160);     //  24,576,000 B (A-swizzled)
  float* XRES   = (float*)(ws + 122972160);   //  45,158,400 B (f32)
  bf16*  YN     = (bf16*)(ws + 168130560);    //  22,579,200 B (A-swizzled)
  bf16*  Y1     = (bf16*)(ws + 190709760);    //  22,579,200 B (row-major)
  float* BIAS   = (float*)(ws + 213288960);   //     518,616 B (head,i,j)
  signed char* CAT = (signed char*)(ws + 213807872);  // 128,000 B (win,tok)
  float* OUT    = (float*)d_out;

  k_prep    <<<180,   256, 0, stream>>>(wqkv, wproj, wpw, WQKVT, WPROJT, WPWT);
  k_tabs    <<<1007,  256, 0, stream>>>(rpb, BIAS, CAT);   // 129654 + 128000 items
  k_ln_shift<<<500,   256, 0, stream>>>(x, g1, b1, XW);
  k_qkv     <<<6000,  256, 0, stream>>>(XW, WQKVT, bqkv, QKV);
  k_attn    <<<24000,  64, 0, stream>>>(QKV, BIAS, CAT, AO);
  k_proj    <<<2000,  256, 0, stream>>>(AO, WPROJT, bproj, x, XRES);
  k_ln2     <<<460,   256, 0, stream>>>(XRES, g2, b2, YN);
  k_pw      <<<1838,  256, 0, stream>>>(YN, WPWT, bpw, Y1);
  k_dw      <<<22050, 256, 0, stream>>>(Y1, wdw, bdw, XRES, OUT);
}
